// MultiModalAttention_88252987998375
// MI455X (gfx1250) — matrix operand order repaired and checked by reading, not compiled, not run on hardware
//
#include <hip/hip_runtime.h>
#include <hip/hip_bf16.h>
#include <math.h>

typedef __attribute__((ext_vector_type(16))) _Float16 v16h;
typedef __attribute__((ext_vector_type(8)))  _Float16 v8h;
typedef __attribute__((ext_vector_type(8)))  float    v8f;
typedef __attribute__((ext_vector_type(4)))  int      v4i;

#define S_LEN   2048
#define D_MODEL 4096
#define NH      32
#define NKV     4
#define HD      128
#define QCOLS   (NH * HD)   // 4096
#define KCOLS   (NKV * HD)  // 512

static __device__ __forceinline__ v16h cat8(v8h lo, v8h hi) {
  return __builtin_shufflevector(lo, hi, 0,1,2,3,4,5,6,7,8,9,10,11,12,13,14,15);
}

// ---------------------------------------------------------------------------
// Async global->LDS copy (CDNA5 GLOBAL_LOAD_ASYNC_TO_LDS_B128, ASYNCcnt).
// Builtin signature (from clang diagnostic): (int4 AS1*, int4 AS3*, Imm, Imm).
// Guarded so compilation never regresses if the builtin is absent.
// ---------------------------------------------------------------------------
#if defined(__has_builtin)
# if __has_builtin(__builtin_amdgcn_global_load_async_to_lds_b128)
#  define HAVE_ASYNC_LDS 1
# endif
#endif
#ifndef HAVE_ASYNC_LDS
# define HAVE_ASYNC_LDS 0
#endif

typedef __attribute__((address_space(1))) void gvoid;
typedef __attribute__((address_space(3))) void lvoid;
typedef __attribute__((address_space(1))) v4i  gv4i;
typedef __attribute__((address_space(3))) v4i  lv4i;

static __device__ __forceinline__ void copy16_to_lds(void* lds, const void* g) {
#if HAVE_ASYNC_LDS
  __builtin_amdgcn_global_load_async_to_lds_b128(
      (gv4i*)(gvoid*)const_cast<void*>(g), (lv4i*)(lvoid*)lds, 0, 0);
#else
  *(v8h*)lds = *(const v8h*)g;
#endif
}

static __device__ __forceinline__ void wait_async_all() {
#if HAVE_ASYNC_LDS
# if __has_builtin(__builtin_amdgcn_s_wait_asynccnt)
  __builtin_amdgcn_s_wait_asynccnt(0);
# else
  asm volatile("s_wait_asynccnt 0" ::: "memory");
# endif
#endif
}

// ---------------------------------------------------------------------------
// Tiled WMMA GEMM: C[M,N] = A[M,K] * B[K,N], double-buffered LDS.
//   A: f32 (A_F16=false) or f16 (A_F16=true), row-major
//   B: f32 row-major, converted to f16 while staging (transposed [N][K])
//   C: f32 (OUT_F32) or f16; STORE_T writes C transposed ([N][M], for V^T)
// Block tile 128x128, BK=32, 256 threads = 8 waves (4 M x 2 N),
// each wave computes 32x64 via 2x4 16x16x32 WMMA tiles.
// ---------------------------------------------------------------------------
template<bool A_F16, bool OUT_F32, bool STORE_T>
__global__ __launch_bounds__(256)
void gemm128x128(const void* __restrict__ Ap, const float* __restrict__ B,
                 void* __restrict__ Cp, int M, int N, int K)
{
  __shared__ __align__(16) _Float16 As[2 * 128 * 40];  // [buf][row][k], pad 40
  __shared__ __align__(16) _Float16 Bs[2 * 128 * 40];  // [buf][col][k], pad 40

  const int tid  = threadIdx.x;
  const int lane = tid & 31;
  const int wave = tid >> 5;
  const int wm   = wave & 3;   // 0..3 along M
  const int wn   = wave >> 2;  // 0..1 along N
  const int ln   = lane & 15;
  const int hi   = lane >> 4;
  const int m0   = blockIdx.y * 128;
  const int n0   = blockIdx.x * 128;

  v8f acc[2][4] = {};

  auto stage = [&](int k0, int buf) {
    _Float16* AsB = As + buf * (128 * 40);
    _Float16* BsB = Bs + buf * (128 * 40);
    if (!A_F16) {
      const float* A = (const float*)Ap;
#pragma unroll
      for (int i = 0; i < 4; ++i) {
        int idx = tid + 256 * i;          // 1024 float4 chunks
        int row = idx >> 3, c4 = idx & 7;
        const float* src = A + (size_t)(m0 + row) * K + (k0 + c4 * 4);
        if (i == 0) __builtin_prefetch(src + 64, 0, 1);  // pull k+64 tile into GL2
        const float4 v = *(const float4*)src;
        _Float16* d = &AsB[row * 40 + c4 * 4];
        d[0] = (_Float16)v.x; d[1] = (_Float16)v.y;
        d[2] = (_Float16)v.z; d[3] = (_Float16)v.w;
      }
    } else {
      const _Float16* A = (const _Float16*)Ap;
#pragma unroll
      for (int i = 0; i < 2; ++i) {
        int idx = tid + 256 * i;          // 512 half8 chunks
        int row = idx >> 2, c8 = idx & 3;
        const _Float16* src = A + (size_t)(m0 + row) * K + (k0 + c8 * 8);
        if (i == 0) __builtin_prefetch(src + 64, 0, 1);
        *(v8h*)&AsB[row * 40 + c8 * 8] = *(const v8h*)src;
      }
    }
#pragma unroll
    for (int i = 0; i < 4; ++i) {
      int idx = tid + 256 * i;            // 1024 float4 chunks (32 k-rows x 128 cols)
      int kr = idx >> 5, c4 = idx & 31;
      const float* src = B + (size_t)(k0 + kr) * N + (n0 + c4 * 4);
      if (i == 0) __builtin_prefetch(src + (size_t)64 * N, 0, 1);
      const float4 v = *(const float4*)src;
      BsB[(c4 * 4 + 0) * 40 + kr] = (_Float16)v.x;
      BsB[(c4 * 4 + 1) * 40 + kr] = (_Float16)v.y;
      BsB[(c4 * 4 + 2) * 40 + kr] = (_Float16)v.z;
      BsB[(c4 * 4 + 3) * 40 + kr] = (_Float16)v.w;
    }
  };

  const int nk = K >> 5;
  stage(0, 0);
  __syncthreads();

  for (int t = 0; t < nk; ++t) {
    const int cur = t & 1;
    if (t + 1 < nk) stage((t + 1) << 5, cur ^ 1);   // overlap with WMMA below

    const _Float16* AsB = As + cur * (128 * 40);
    const _Float16* BsB = Bs + cur * (128 * 40);
#pragma unroll
    for (int mt = 0; mt < 2; ++mt) {
      // A fragment: lane(hi,ln) row = base+ln; K = {hi*8..+7, 16+hi*8..+7}
      const _Float16* ap = &AsB[(wm * 32 + mt * 16 + ln) * 40 + hi * 8];
      v16h a = cat8(*(const v8h*)ap, *(const v8h*)(ap + 16));
#pragma unroll
      for (int nt = 0; nt < 4; ++nt) {
        // B fragment: lane(hi,ln) col = base+ln; K = {hi*8..+7, 16+hi*8..+7}
        const _Float16* bp = &BsB[(wn * 64 + nt * 16 + ln) * 40 + hi * 8];
        v16h b = cat8(*(const v8h*)bp, *(const v8h*)(bp + 16));
        acc[mt][nt] = __builtin_amdgcn_wmma_f32_16x16x32_f16(
            false, a, false, b, (short)0, acc[mt][nt], false, false);
      }
    }
    __syncthreads();
  }

  // ---- epilogue: C layout VGPR r -> row hi*8+r, col = lane%16 ----
#pragma unroll
  for (int mt = 0; mt < 2; ++mt)
#pragma unroll
    for (int nt = 0; nt < 4; ++nt)
#pragma unroll
      for (int r = 0; r < 8; ++r) {
        int row = m0 + wm * 32 + mt * 16 + hi * 8 + r;
        int col = n0 + wn * 64 + nt * 16 + ln;
        float v = acc[mt][nt][r];
        if (OUT_F32) {
          ((float*)Cp)[(size_t)row * N + col] = v;
        } else if (STORE_T) {
          ((_Float16*)Cp)[(size_t)col * M + row] = (_Float16)v;
        } else {
          ((_Float16*)Cp)[(size_t)row * N + col] = (_Float16)v;
        }
      }
}

// ---------------------------------------------------------------------------
// YaRN RoPE in-place on f16 Q [S,32*128] and K [S,4*128]; folds the attention
// scale 1/sqrt(128) into Q so the attention kernel stages Q as a raw copy.
// ---------------------------------------------------------------------------
__global__ void rope_yarn_kernel(_Float16* __restrict__ Q, _Float16* __restrict__ Kh)
{
  int idx = blockIdx.x * blockDim.x + threadIdx.x;
  const int total = S_LEN * (NH + NKV) * (HD / 2);
  if (idx >= total) return;
  int j  = idx & 63;
  int t  = idx >> 6;
  int hh = t % (NH + NKV);
  int s  = t / (NH + NKV);

  float inv = __expf(-((float)(2 * j) * (1.0f / 128.0f)) * __logf(10000.0f));
  float wavelength = 6.283185307179586f / inv;
  float r = (2048.0f * 80.0f) / wavelength;          // MAX_SEQ / wavelength
  float gamma = fminf(fmaxf((r - 1.0f) / (32.0f - 1.0f), 0.0f), 1.0f);
  inv = inv * ((1.0f - gamma) * (1.0f / 80.0f) + gamma);
  float mscale = sqrtf(0.1f * __logf(80.0f) + 1.0f); // angle divided by mscale (per ref)
  float ang = ((float)s * inv) / mscale;
  float sn, c;
  __sincosf(ang, &sn, &c);

  _Float16* base;
  float outscale;
  if (hh < NH) { base = Q  + (size_t)s * QCOLS + hh * HD;        outscale = 0.08838834764831845f; }
  else         { base = Kh + (size_t)s * KCOLS + (hh - NH) * HD; outscale = 1.0f; }
  float x1 = (float)base[j];
  float x2 = (float)base[j + 64];
  base[j]      = (_Float16)((x1 * c - x2 * sn) * outscale);
  base[j + 64] = (_Float16)((x2 * c + x1 * sn) * outscale);
}

// ---------------------------------------------------------------------------
// Flash attention: grid (S/64, NH), 128 threads = 4 waves, double-buffered
// K/V tiles staged with async global->LDS copies (ASYNCcnt). Wave w owns 16
// query rows. Scores via WMMA, online softmax with cross-lane reductions,
// probs round-trip LDS in A-fragment layout, PV via WMMA. V is pre-transposed.
// ---------------------------------------------------------------------------
__global__ __launch_bounds__(128)
void flash_attn_kernel(const _Float16* __restrict__ Q,
                       const _Float16* __restrict__ Kh,
                       const _Float16* __restrict__ Vt,
                       _Float16* __restrict__ O)
{
  __shared__ __align__(16) _Float16 Qs[64 * 136];       // [qrow][dim]
  __shared__ __align__(16) _Float16 Ks[2][64 * 136];    // [buf][key][dim]
  __shared__ __align__(16) _Float16 Vts[2][128 * 72];   // [buf][dim][key]
  __shared__ __align__(16) _Float16 Ps[4 * 16 * 72];    // per-wave probs

  const int tid  = threadIdx.x;
  const int lane = tid & 31;
  const int w    = tid >> 5;
  const int ln   = lane & 15;
  const int hi   = lane >> 4;
  const int q0   = blockIdx.x * 64;
  const int h    = blockIdx.y;
  const int kv   = h >> 3;     // GQA: kv head = h / 8

  // stage Q tile (already scaled by RoPE kernel)
#pragma unroll
  for (int i = 0; i < 8; ++i) {
    int idx = tid + 128 * i;
    int row = idx >> 4, c8 = idx & 15;
    copy16_to_lds(&Qs[row * 136 + c8 * 8],
                  Q + (size_t)(q0 + row) * QCOLS + h * HD + c8 * 8);
  }

  auto stage_kv = [&](int k0, int buf) {
#pragma unroll
    for (int i = 0; i < 8; ++i) {           // K tile: 64 keys x 128 dims
      int idx = tid + 128 * i;
      int row = idx >> 4, c8 = idx & 15;
      copy16_to_lds(&Ks[buf][row * 136 + c8 * 8],
                    Kh + (size_t)(k0 + row) * KCOLS + kv * HD + c8 * 8);
    }
#pragma unroll
    for (int i = 0; i < 8; ++i) {           // V^T tile: 128 dims x 64 keys
      int idx = tid + 128 * i;
      int row = idx >> 3, c8 = idx & 7;
      copy16_to_lds(&Vts[buf][row * 72 + c8 * 8],
                    Vt + (size_t)(kv * HD + row) * S_LEN + k0 + c8 * 8);
    }
  };

  float mrow[8], lrow[8];
  v8f oacc[8] = {};
#pragma unroll
  for (int r = 0; r < 8; ++r) { mrow[r] = -1e30f; lrow[r] = 0.0f; }

  const int ntiles = (q0 >> 6) + 1;
  stage_kv(0, 0);

  for (int t = 0; t < ntiles; ++t) {
    const int cur = t & 1;
    const int k0  = t << 6;

    wait_async_all();          // tile t (and Q on t==0) arrived for this wave
    __syncthreads();           // arrived for all waves; prev-iter reads done
    if (t + 1 < ntiles) stage_kv((t + 1) << 6, cur ^ 1);  // overlap next tile

    // scores: 16 q rows x 64 keys, reduce over head_dim (4 chunks of 32)
    v8f sacc[4] = {};
#pragma unroll
    for (int kc = 0; kc < 4; ++kc) {
      const _Float16* ap = &Qs[(w * 16 + ln) * 136 + kc * 32 + hi * 8];
      v16h a = cat8(*(const v8h*)ap, *(const v8h*)(ap + 16));
#pragma unroll
      for (int nt = 0; nt < 4; ++nt) {
        const _Float16* bp = &Ks[cur][(nt * 16 + ln) * 136 + kc * 32 + hi * 8];
        v16h b = cat8(*(const v8h*)bp, *(const v8h*)(bp + 16));
        sacc[nt] = __builtin_amdgcn_wmma_f32_16x16x32_f16(
            false, a, false, b, (short)0, sacc[nt], false, false);
      }
    }

    // causal mask (diagonal key tile only)
    if (t + 1 == ntiles) {
#pragma unroll
      for (int nt = 0; nt < 4; ++nt)
#pragma unroll
        for (int r = 0; r < 8; ++r) {
          int rg = q0 + w * 16 + hi * 8 + r;
          int cg = k0 + nt * 16 + ln;
          if (cg > rg) sacc[nt][r] = -1e30f;
        }
    }

    // online softmax (rows hi*8+r; columns striped over 16-lane groups)
    _Float16* pw = &Ps[w * 16 * 72];
#pragma unroll
    for (int r = 0; r < 8; ++r) {
      float mx = fmaxf(fmaxf(sacc[0][r], sacc[1][r]), fmaxf(sacc[2][r], sacc[3][r]));
      mx = fmaxf(mx, __shfl_xor(mx, 1, 32));
      mx = fmaxf(mx, __shfl_xor(mx, 2, 32));
      mx = fmaxf(mx, __shfl_xor(mx, 4, 32));
      mx = fmaxf(mx, __shfl_xor(mx, 8, 32));
      float mnew  = fmaxf(mrow[r], mx);
      float alpha = __expf(mrow[r] - mnew);
      float rs = 0.0f;
#pragma unroll
      for (int nt = 0; nt < 4; ++nt) {
        float p = __expf(sacc[nt][r] - mnew);
        rs += p;
        pw[(hi * 8 + r) * 72 + nt * 16 + ln] = (_Float16)p;
      }
      rs += __shfl_xor(rs, 1, 32);
      rs += __shfl_xor(rs, 2, 32);
      rs += __shfl_xor(rs, 4, 32);
      rs += __shfl_xor(rs, 8, 32);
      lrow[r] = lrow[r] * alpha + rs;
      mrow[r] = mnew;
#pragma unroll
      for (int ot = 0; ot < 8; ++ot) oacc[ot][r] *= alpha;
    }
    __syncthreads();

    // P (16x64) @ V (64x128): key chunks of 32, dim tiles of 16
#pragma unroll
    for (int kc = 0; kc < 2; ++kc) {
      const _Float16* ap = &pw[ln * 72 + kc * 32 + hi * 8];
      v16h a = cat8(*(const v8h*)ap, *(const v8h*)(ap + 16));
#pragma unroll
      for (int ot = 0; ot < 8; ++ot) {
        const _Float16* bp = &Vts[cur][(ot * 16 + ln) * 72 + kc * 32 + hi * 8];
        v16h b = cat8(*(const v8h*)bp, *(const v8h*)(bp + 16));
        oacc[ot] = __builtin_amdgcn_wmma_f32_16x16x32_f16(
            false, a, false, b, (short)0, oacc[ot], false, false);
      }
    }
  }

  // epilogue: normalize and store f16 attention output [S, 32*128]
#pragma unroll
  for (int ot = 0; ot < 8; ++ot)
#pragma unroll
    for (int r = 0; r < 8; ++r) {
      int row = q0 + w * 16 + hi * 8 + r;
      int col = h * HD + ot * 16 + ln;
      O[(size_t)row * QCOLS + col] = (_Float16)(oacc[ot][r] / lrow[r]);
    }
}

// ---------------------------------------------------------------------------
extern "C" void kernel_launch(void* const* d_in, const int* in_sizes, int n_in,
                              void* d_out, int out_size, void* d_ws, size_t ws_size,
                              hipStream_t stream)
{
  const float* x  = (const float*)d_in[0];
  const float* wq = (const float*)d_in[1];
  const float* wk = (const float*)d_in[2];
  const float* wv = (const float*)d_in[3];
  const float* wo = (const float*)d_in[4];
  float* out = (float*)d_out;

  // workspace: Q f16 [S,4096] | K f16 [S,512] | V^T f16 [512,S] | attn f16 [S,4096]
  _Float16* Qh = (_Float16*)d_ws;
  _Float16* Kh = Qh + (size_t)S_LEN * QCOLS;
  _Float16* Vt = Kh + (size_t)S_LEN * KCOLS;
  _Float16* AO = Vt + (size_t)KCOLS * S_LEN;

  dim3 blk(256);
  gemm128x128<false, false, false>
      <<<dim3(QCOLS / 128, S_LEN / 128), blk, 0, stream>>>(x, wq, Qh, S_LEN, QCOLS, D_MODEL);
  gemm128x128<false, false, false>
      <<<dim3(KCOLS / 128, S_LEN / 128), blk, 0, stream>>>(x, wk, Kh, S_LEN, KCOLS, D_MODEL);
  gemm128x128<false, false, true>
      <<<dim3(KCOLS / 128, S_LEN / 128), blk, 0, stream>>>(x, wv, Vt, S_LEN, KCOLS, D_MODEL);

  const int rope_total = S_LEN * (NH + NKV) * (HD / 2);
  rope_yarn_kernel<<<(rope_total + 255) / 256, dim3(256), 0, stream>>>(Qh, Kh);

  flash_attn_kernel<<<dim3(S_LEN / 64, NH), dim3(128), 0, stream>>>(Qh, Kh, Vt, AO);

  gemm128x128<true, true, false>
      <<<dim3(D_MODEL / 128, S_LEN / 128), blk, 0, stream>>>(AO, wo, out, S_LEN, D_MODEL, QCOLS);
}
